// AlignModule_full_13520557047840
// MI455X (gfx1250) — matrix operand order repaired and checked by reading, not compiled, not run on hardware
//
#include <hip/hip_runtime.h>

typedef _Float16 v16h __attribute__((ext_vector_type(16)));
typedef _Float16 h8   __attribute__((ext_vector_type(8)));
typedef float    v8f  __attribute__((ext_vector_type(8)));

#define HW_  32768
#define W_   256
#define H_   128

// ---------------------------------------------------------------------------
// Kernel 0: repack w_flow1 (64,128,3,3) OIHW f32 -> wBt (64 c, 1152 kk) f16
// kk = tap*128 + ch so B-matrix fragment loads are contiguous 16B chunks.
// ---------------------------------------------------------------------------
__global__ void __launch_bounds__(256) k_pack(const float* __restrict__ wf1,
                                              _Float16* __restrict__ wBt) {
    int i = blockIdx.x * 256 + threadIdx.x;   // 64*1152 = 73728
    if (i >= 64 * 1152) return;
    int c   = i / 1152;
    int kk  = i % 1152;
    int tap = kk >> 7;
    int ch  = kk & 127;
    wBt[i] = (_Float16)wf1[(c * 128 + ch) * 9 + tap];
}

// ---------------------------------------------------------------------------
// Kernel 1: fused down1/down2 1x1-conv GEMM, f32->f16 convert, f32 accum WMMA.
// Output xcat: (N, HW, 128) f16, pixel-major (concat of t1-proj | t2-proj).
// HBM-bound (268 MB fp32 feature reads) -> direct coalesced loads, no staging.
// grid (HW/32, N, 2src), block 256 (8 waves: 4 ctiles x 2 pixel subtiles).
// ---------------------------------------------------------------------------
__global__ void __launch_bounds__(256) k_down(const float* __restrict__ t1,
                                              const float* __restrict__ t2,
                                              const float* __restrict__ w1,
                                              const float* __restrict__ w2,
                                              _Float16* __restrict__ xcat) {
    int lane  = threadIdx.x & 31;
    int wave  = threadIdx.x >> 5;
    int ctile = wave & 3;
    int psub  = wave >> 2;
    int src   = blockIdx.z;
    int n     = blockIdx.y;
    int P0    = blockIdx.x * 32 + psub * 16;

    const float* feat = (src == 0) ? t1 : t2;
    const float* w    = (src == 0) ? w1 : w2;

    int m = lane & 15;   // A: row M / B: col N / D: N-column
    int h = lane >> 4;   // K-half selector

    int p  = P0 + m;
    int cN = ctile * 16 + m;

    const float* featp = feat + (size_t)n * 256 * HW_ + p;  // + k*HW
    const float* wp    = w + (size_t)cN * 256;              // + k

    v8f acc = {};
    #pragma unroll
    for (int ks = 0; ks < 256; ks += 32) {
        v16h a, b;
        #pragma unroll
        for (int e = 0; e < 16; ++e) {
            // A (16x32 f16): k = (e&7) + 8h + 16*(e>>3)
            int ka = ks + (e & 7) + (h << 3) + ((e >> 3) << 4);
            a[e] = (_Float16)featp[(size_t)ka * HW_];
            // B (32x16 f16): k = (e&7) + 8h + 16*(e>>3)
            int kb = ks + (e & 7) + (h << 3) + ((e >> 3) << 4);
            b[e] = (_Float16)wp[kb];
        }
        acc = __builtin_amdgcn_wmma_f32_16x16x32_f16(false, a, false, b,
                                                     (short)0, acc, false, false);
    }
    // D layout: lane l, vgpr r -> M = r + 8h, N = l&15
    #pragma unroll
    for (int r = 0; r < 8; ++r) {
        int pr = P0 + r + (h << 3);
        xcat[((size_t)(n * HW_ + pr)) * 128 + src * 64 + ctile * 16 + m] =
            (_Float16)acc[r];
    }
}

// ---------------------------------------------------------------------------
// Kernel 2: 3x3 conv (128->64) as im2col WMMA GEMM (K = 9*128), fused BN+ReLU.
// Compute-bound -> stage the block's (3 x 34 x 128ch) f16 halo tile in LDS
// once via gfx1250 async copy (global_load_async_to_lds_b128 / ASYNCcnt),
// zero-filling OOB halo so the WMMA loop needs no predication. A-fragments
// then come from LDS (ds_load_b128), shared by all 8 waves and all 9 taps.
// grid (HW/32, N), block 256.
// ---------------------------------------------------------------------------
__global__ void __launch_bounds__(256) k_conv1(const _Float16* __restrict__ xcat,
                                               const _Float16* __restrict__ wBt,
                                               const float* __restrict__ bn_g,
                                               const float* __restrict__ bn_b,
                                               const float* __restrict__ bn_m,
                                               const float* __restrict__ bn_v,
                                               _Float16* __restrict__ y) {
    __shared__ __attribute__((aligned(16))) _Float16 sm[3 * 34 * 128]; // 25.5 KB

    int lane  = threadIdx.x & 31;
    int wave  = threadIdx.x >> 5;
    int ctile = wave & 3;
    int psub  = wave >> 2;
    int n     = blockIdx.y;
    int Pblk  = blockIdx.x * 32;          // 32 consecutive pixels, one image row
    int P0    = Pblk + psub * 16;

    int py  = Pblk >> 8;                  // image row of the tile
    int px0 = Pblk & (W_ - 1);            // first column of the tile

    // ---- async-stage halo tile: rows py-1..py+1, cols px0-1..px0+32 ----
    // 102 pixel-vectors x 128ch f16 = 1632 chunks of 16B.
    for (int i = threadIdx.x; i < 1632; i += 256) {
        int pixel = i >> 4;               // 0..101
        int chunk = i & 15;               // 16B chunk within 128-ch vector
        int r  = pixel / 34;
        int j  = pixel - r * 34;
        int gy = py + r - 1;
        int gx = px0 + j - 1;
        int half_off = pixel * 128 + chunk * 8;
        if (gy >= 0 && gy < H_ && gx >= 0 && gx < W_) {
            unsigned long long gaddr = (unsigned long long)
                (xcat + ((size_t)(n * HW_ + gy * W_ + gx)) * 128 + chunk * 8);
            unsigned lds_off = (unsigned)(unsigned long long)&sm[half_off];
            asm volatile("global_load_async_to_lds_b128 %0, %1, off"
                         :: "v"(lds_off), "v"(gaddr) : "memory");
        } else {
            h8 z = {};
            *(h8*)&sm[half_off] = z;      // zero padding halo
        }
    }
    asm volatile("s_wait_asynccnt 0x0" ::: "memory");
    __syncthreads();

    int m = lane & 15;
    int h = lane >> 4;
    int jl = psub * 16 + m;               // local column (before +dx+1)
    int cN = ctile * 16 + m;

    const h8* wrow = (const h8*)(wBt + (size_t)cN * 1152);
    v8f acc = {};

    #pragma unroll
    for (int tap = 0; tap < 9; ++tap) {
        const int r   = tap / 3;          // dy + 1
        const int dxi = tap % 3;          // dx + 1
        int pix = r * 34 + jl + dxi;      // staged pixel index
        #pragma unroll
        for (int ks2 = 0; ks2 < 4; ++ks2) {
            int base = tap * 128 + ks2 * 32;                 // B k-offset (halfs)
            h8 a0 = *(const h8*)&sm[pix * 128 + ks2 * 32 + h * 8];       // k=(e&7)+8h
            h8 a1 = *(const h8*)&sm[pix * 128 + ks2 * 32 + 16 + h * 8];  // k=+16
            h8 b0 = wrow[(base + h * 8) >> 3];               // k=(e&7)+8h
            h8 b1 = wrow[(base + 16 + h * 8) >> 3];
            v16h a = __builtin_shufflevector(a0, a1, 0,1,2,3,4,5,6,7,8,9,10,11,12,13,14,15);
            v16h b = __builtin_shufflevector(b0, b1, 0,1,2,3,4,5,6,7,8,9,10,11,12,13,14,15);
            acc = __builtin_amdgcn_wmma_f32_16x16x32_f16(false, a, false, b,
                                                         (short)0, acc, false, false);
        }
    }

    float sc = bn_g[cN] * rsqrtf(bn_v[cN] + 1e-5f);
    float mu = bn_m[cN], bb = bn_b[cN];
    #pragma unroll
    for (int r = 0; r < 8; ++r) {
        int pr = P0 + r + (h << 3);
        float v = (acc[r] - mu) * sc + bb;
        v = v > 0.f ? v : 0.f;
        y[((size_t)(n * HW_ + pr)) * 64 + cN] = (_Float16)v;
    }
}

// ---------------------------------------------------------------------------
// Kernel 3: 3x3 conv (64 -> 2 flow components). Only 2 out channels -> VALU
// with LDS-staged weights. y is pixel-major so channel reads are contiguous.
// grid (HW/256, N), block 256. flow ws layout: (N, 2, HW) f32.
// ---------------------------------------------------------------------------
__global__ void __launch_bounds__(256) k_flow(const _Float16* __restrict__ y,
                                              const float* __restrict__ wf2,
                                              float* __restrict__ flow) {
    __shared__ float wl[1152];   // [comp][tap][ch]
    for (int i = threadIdx.x; i < 1152; i += 256) {
        int comp = i / 576, rem = i % 576, tap = rem >> 6, ch = rem & 63;
        wl[i] = wf2[(comp * 64 + ch) * 9 + tap];
    }
    __syncthreads();

    int p  = blockIdx.x * 256 + threadIdx.x;
    int n  = blockIdx.y;
    int px = p & (W_ - 1);
    int py = p >> 8;

    float a0 = 0.f, a1 = 0.f;
    for (int tap = 0; tap < 9; ++tap) {
        int dy = tap / 3 - 1, dx = tap % 3 - 1;
        int qy = py + dy, qx = px + dx;
        if (qy < 0 || qy >= H_ || qx < 0 || qx >= W_) continue;
        const _Float16* yp = y + ((size_t)(n * HW_ + qy * W_ + qx)) * 64;
        #pragma unroll
        for (int ch = 0; ch < 64; ++ch) {
            float v = (float)yp[ch];
            a0 += v * wl[tap * 64 + ch];
            a1 += v * wl[576 + tap * 64 + ch];
        }
    }
    flow[(size_t)(n * 2 + 0) * HW_ + p] = a0;
    flow[(size_t)(n * 2 + 1) * HW_ + p] = a1;
}

// ---------------------------------------------------------------------------
// Kernel 4: flow_warp with faithful torch repeat semantics:
// image (n,c) uses flow[(n*19 + c) % 4]; bilinear, align_corners=False,
// zero padding. One thread per output element.
// ---------------------------------------------------------------------------
__global__ void __launch_bounds__(256) k_warp(const float* __restrict__ pred,
                                              const float* __restrict__ flow,
                                              float* __restrict__ out) {
    const int C = 19;
    int idx = blockIdx.x * 256 + threadIdx.x;
    if (idx >= 4 * C * HW_) return;
    int p  = idx % HW_;
    int t  = idx / HW_;
    int cc = t % C, n = t / C;
    int px = p & (W_ - 1);
    int py = p >> 8;

    int fsel = (n * C + cc) & 3;  // (n*c + c_i) % n, n == 4
    float fx = flow[(size_t)(fsel * 2 + 0) * HW_ + p];
    float fy = flow[(size_t)(fsel * 2 + 1) * HW_ + p];

    float gx = -1.f + 2.f * px / (float)(W_ - 1);
    float gy = -1.f + 2.f * py / (float)(H_ - 1);
    float sx = gx + fx / (float)W_;
    float sy = gy + fy / (float)H_;
    float ix = ((sx + 1.f) * W_ - 1.f) * 0.5f;
    float iy = ((sy + 1.f) * H_ - 1.f) * 0.5f;

    float x0f = floorf(ix), y0f = floorf(iy);
    float wx1 = ix - x0f, wy1 = iy - y0f;
    int x0 = (int)x0f, y0 = (int)y0f;

    const float* img = pred + (size_t)(n * C + cc) * HW_;
    auto G = [&](int yy, int xx) -> float {
        if (xx < 0 || xx >= W_ || yy < 0 || yy >= H_) return 0.f;
        return img[yy * W_ + xx];
    };
    float v00 = G(y0, x0),     v01 = G(y0, x0 + 1);
    float v10 = G(y0 + 1, x0), v11 = G(y0 + 1, x0 + 1);
    float top = v00 * (1.f - wx1) + v01 * wx1;
    float bot = v10 * (1.f - wx1) + v11 * wx1;
    out[idx] = top * (1.f - wy1) + bot * wy1;
}

// ---------------------------------------------------------------------------
extern "C" void kernel_launch(void* const* d_in, const int* in_sizes, int n_in,
                              void* d_out, int out_size, void* d_ws, size_t ws_size,
                              hipStream_t stream) {
    const float* t1f  = (const float*)d_in[0];
    const float* t2f  = (const float*)d_in[1];
    const float* pred = (const float*)d_in[2];
    const float* wd1  = (const float*)d_in[3];
    const float* wd2  = (const float*)d_in[4];
    const float* wf1  = (const float*)d_in[5];
    const float* bng  = (const float*)d_in[6];
    const float* bnb  = (const float*)d_in[7];
    const float* bnm  = (const float*)d_in[8];
    const float* bnv  = (const float*)d_in[9];
    const float* wf2  = (const float*)d_in[10];
    float* out = (float*)d_out;

    char* ws = (char*)d_ws;
    const size_t XCAT_B = (size_t)4 * HW_ * 128 * 2;  // 32 MB
    const size_t Y_B    = (size_t)4 * HW_ * 64 * 2;   // 16 MB
    const size_t FLOW_B = (size_t)4 * 2 * HW_ * 4;    // 1 MB
    _Float16* xcat = (_Float16*)ws;
    _Float16* yb   = (_Float16*)(ws + XCAT_B);
    float*    flow = (float*)(ws + XCAT_B + Y_B);
    _Float16* wBt  = (_Float16*)(ws + XCAT_B + Y_B + FLOW_B);

    k_pack <<<288, 256, 0, stream>>>(wf1, wBt);
    k_down <<<dim3(HW_ / 32, 4, 2), 256, 0, stream>>>(t1f, t2f, wd1, wd2, xcat);
    k_conv1<<<dim3(HW_ / 32, 4),    256, 0, stream>>>(xcat, wBt, bng, bnb, bnm, bnv, yb);
    k_flow <<<dim3(HW_ / 256, 4),   256, 0, stream>>>(yb, wf2, flow);
    k_warp <<<(4 * 19 * HW_) / 256, 256, 0, stream>>>(pred, flow, out);
}
